// Encoder_24618752540919
// MI455X (gfx1250) — matrix operand order repaired and checked by reading, not compiled, not run on hardware
//
#include <hip/hip_runtime.h>
#include <stdint.h>

// ---------------------------------------------------------------------------
// Types for CDNA5 WMMA (wave32): v_wmma_f32_16x16x32_bf16
// ---------------------------------------------------------------------------
typedef __attribute__((ext_vector_type(16))) __bf16 v16bf;
typedef __attribute__((ext_vector_type(8)))  float  v8f;

// ---------------------------------------------------------------------------
// Async global->LDS copy (CDNA5, tracked by ASYNCcnt) + waits
// ---------------------------------------------------------------------------
__device__ __forceinline__ uint32_t lds_off32(const void* p) {
    // generic pointer to LDS: low 32 bits are the workgroup-relative LDS offset
    return (uint32_t)(uintptr_t)p;
}
__device__ __forceinline__ void async_copy_b128(uint32_t lds_off, const void* gptr) {
    asm volatile("global_load_async_to_lds_b128 %0, %1, off"
                 :: "v"(lds_off), "v"(gptr) : "memory");
}
__device__ __forceinline__ void wait_async_le0()  { asm volatile("s_wait_asynccnt 0x0"  ::: "memory"); }
__device__ __forceinline__ void wait_async_le6()  { asm volatile("s_wait_asynccnt 0x6"  ::: "memory"); }
__device__ __forceinline__ void wait_async_le16() { asm volatile("s_wait_asynccnt 0x10" ::: "memory"); }
__device__ __forceinline__ void wave_fence() {
    __builtin_amdgcn_fence(__ATOMIC_ACQ_REL, "wavefront");
}

// ---------------------------------------------------------------------------
// bf16 helpers (stored as raw unsigned short)
// ---------------------------------------------------------------------------
__device__ __forceinline__ unsigned short f32_to_bf16(float f) {
    union { float f; uint32_t u; } c;
    c.f = f;
    uint32_t u = c.u;
    u += 0x7FFFu + ((u >> 16) & 1u);   // round-to-nearest-even
    return (unsigned short)(u >> 16);
}

// A-fragment (16x32 bf16, M x K), row-major source, row stride rs (elements).
// ISA: lanes 0-15 hold M=lane; K = kb+e (e=0..7), 16+kb+e (e=8..15), kb=(lane>>4)*8.
__device__ __forceinline__ v16bf load_frag_a(const unsigned short* p, int rs, int lane) {
    int row = lane & 15;
    int kb  = (lane >> 4) << 3;
    union { uint4 q[2]; v16bf v; } u;
    const unsigned short* base = p + (size_t)row * rs;
    u.q[0] = *(const uint4*)(base + kb);
    u.q[1] = *(const uint4*)(base + 16 + kb);
    return u.v;
}

// B-fragment (32x16 bf16, K x N) from TRANSPOSED storage pT[N][K], row stride rs.
// ISA: lane holds N=lane&15; K = kb+e (e=0..7), 16+kb+e (e=8..15), kb=(lane>>4)*8.
__device__ __forceinline__ v16bf load_frag_b(const unsigned short* pT, int rs, int lane) {
    int col = lane & 15;
    int kb  = (lane >> 4) << 3;
    union { uint4 q[2]; v16bf v; } u;
    const unsigned short* base = pT + (size_t)col * rs + kb;
    u.q[0] = *(const uint4*)(base);
    u.q[1] = *(const uint4*)(base + 16);
    return u.v;
}

// ---------------------------------------------------------------------------
// Elementwise f32 -> bf16 (4 per thread)
// ---------------------------------------------------------------------------
__global__ __launch_bounds__(256) void cvt_bf16_kernel(const float* __restrict__ in,
                                                       unsigned short* __restrict__ out,
                                                       int n) {
    int i = (blockIdx.x * 256 + threadIdx.x) * 4;
    if (i >= n) return;
    float4 v = *(const float4*)(in + i);
    uint2 o;
    o.x = (uint32_t)f32_to_bf16(v.x) | ((uint32_t)f32_to_bf16(v.y) << 16);
    o.y = (uint32_t)f32_to_bf16(v.z) | ((uint32_t)f32_to_bf16(v.w) << 16);
    *(uint2*)(out + i) = o;
}

// ---------------------------------------------------------------------------
// Weight transpose + convert: WT[n*K + k] = bf16(W[k*N + n])
// ---------------------------------------------------------------------------
__global__ __launch_bounds__(256) void transpose_cvt_kernel(const float* __restrict__ W,
                                                            unsigned short* __restrict__ WT,
                                                            int K, int N) {
    size_t idx = (size_t)blockIdx.x * 256 + threadIdx.x;
    if (idx >= (size_t)K * N) return;
    int n = (int)(idx / K);
    int k = (int)(idx % K);
    WT[idx] = f32_to_bf16(W[(size_t)k * N + n]);
}

// ---------------------------------------------------------------------------
// V transpose: Vp [B,S,D] (head-major cols) -> Vt [B,H,hd,S]
// ---------------------------------------------------------------------------
__global__ __launch_bounds__(256) void transpose_v_kernel(const unsigned short* __restrict__ Vp,
                                                          unsigned short* __restrict__ Vt,
                                                          int B, int S, int D, int H, int hd) {
    size_t idx = (size_t)blockIdx.x * 256 + threadIdx.x;
    if (idx >= (size_t)B * S * D) return;
    int c = (int)(idx % D);
    size_t bs = idx / D;
    int s = (int)(bs % S);
    int b = (int)(bs / S);
    int h = c / hd;
    int dd = c % hd;
    Vt[(((size_t)(b * H + h) * hd) + dd) * S + s] = Vp[idx];
}

// ---------------------------------------------------------------------------
// Tiled bf16 WMMA GEMM with async-LDS double-buffered pipeline.
//   A  : [M x K] bf16 row-major ; BT : [N x K] bf16 (B transposed)
// Block tile 256(M) x 128(N), BK=32; 256 threads = 8 waves (4 M x 2 N);
// wave tile 64x64 = 16 accumulators -> 16 wmma per 8 fragment loads per step.
// Staging: 6 x global_load_async_to_lds_b128 per thread per K-step.
// ---------------------------------------------------------------------------
template <bool RELU, bool OUT_F32, bool OUT_BF16>
__global__ __launch_bounds__(256) void gemm_bf16_wmma_kernel(
    const unsigned short* __restrict__ A,
    const unsigned short* __restrict__ BT,
    const float* __restrict__ bias,
    float* __restrict__ Cf,
    unsigned short* __restrict__ Cb,
    int M, int N, int K) {

    __shared__ unsigned short Al[2][256 * 40];   // stride 40 (80B rows): 16B aligned
    __shared__ unsigned short Bl[2][128 * 40];

    const int tid  = threadIdx.x;
    const int lane = tid & 31;
    const int wave = tid >> 5;
    const int wm   = wave >> 1;   // 0..3 -> 64-row slab
    const int wn   = wave & 1;    // 0..1 -> 64-col slab
    const int n0   = blockIdx.x * 128;
    const int m0   = blockIdx.y * 256;

    // stage one K-tile: A = 1024 16B-chunks, B = 512 16B-chunks -> 4+2 per thread
    auto stage = [&](int buf, int k0) {
#pragma unroll
        for (int c = tid; c < 1024; c += 256) {
            int row = c >> 2, part = c & 3;
            async_copy_b128(lds_off32(&Al[buf][row * 40 + part * 8]),
                            A + (size_t)(m0 + row) * K + k0 + part * 8);
        }
#pragma unroll
        for (int c = tid; c < 512; c += 256) {
            int row = c >> 2, part = c & 3;
            async_copy_b128(lds_off32(&Bl[buf][row * 40 + part * 8]),
                            BT + (size_t)(n0 + row) * K + k0 + part * 8);
        }
    };

    v8f acc[4][4];
#pragma unroll
    for (int a = 0; a < 4; ++a)
#pragma unroll
        for (int b = 0; b < 4; ++b) acc[a][b] = {};

    const int KT = K / 32;
    stage(0, 0);

    for (int kt = 0; kt < KT; ++kt) {
        const int cur = kt & 1;
        if (kt + 1 < KT) {
            stage(cur ^ 1, (kt + 1) * 32);
            wait_async_le6();   // in-order completion -> current tile landed
        } else {
            wait_async_le0();
        }
        __syncthreads();

        // batch all fragment loads first (distinct registers -> counted waits,
        // loads clause together instead of wait_dscnt0 per wmma)
        v16bf af[4], bfg[4];
#pragma unroll
        for (int mf = 0; mf < 4; ++mf)
            af[mf] = load_frag_a(&Al[cur][(wm * 64 + mf * 16) * 40], 40, lane);
#pragma unroll
        for (int nf = 0; nf < 4; ++nf)
            bfg[nf] = load_frag_b(&Bl[cur][(wn * 64 + nf * 16) * 40], 40, lane);
#pragma unroll
        for (int mf = 0; mf < 4; ++mf)
#pragma unroll
            for (int nf = 0; nf < 4; ++nf)
                acc[mf][nf] = __builtin_amdgcn_wmma_f32_16x16x32_bf16(
                    false, af[mf], false, bfg[nf], (short)0, acc[mf][nf], false, false);
        __syncthreads();   // all waves done reading buf[cur] before it is overwritten
    }

    // epilogue: bias + (relu) + store
    const int half = lane >> 4;
    const int lcol = lane & 15;
#pragma unroll
    for (int nf = 0; nf < 4; ++nf) {
        int gcol = n0 + wn * 64 + nf * 16 + lcol;
        float bv = bias ? bias[gcol] : 0.0f;
#pragma unroll
        for (int mf = 0; mf < 4; ++mf) {
#pragma unroll
            for (int i = 0; i < 8; ++i) {
                int grow = m0 + wm * 64 + mf * 16 + i + 8 * half;
                float v = acc[mf][nf][i] + bv;
                if (RELU) v = fmaxf(v, 0.0f);
                if (OUT_F32)  Cf[(size_t)grow * N + gcol] = v;
                if (OUT_BF16) Cb[(size_t)grow * N + gcol] = f32_to_bf16(v);
            }
        }
    }
}

// ---------------------------------------------------------------------------
// Flash attention, hd=128, bf16 WMMA, fp32 online softmax.
// grid: (S/64, H, B), block 128 (4 waves); each wave owns 16 query rows.
// K/V key-tiles (64 keys) staged in LDS, shared by all 4 waves, double-buffered
// via async copies (16 per thread per stage).
// Qp,Kp: [B,S,D] bf16 head-major cols; Vt: [B,H,hd,S] bf16; mh: [B,S,D] f32.
// ---------------------------------------------------------------------------
__global__ __launch_bounds__(128) void attn_flash_kernel(
    const unsigned short* __restrict__ Qp,
    const unsigned short* __restrict__ Kp,
    const unsigned short* __restrict__ Vt,
    float* __restrict__ mh,
    int S, int D, int H) {

    const int hd   = 128;
    const int tid  = threadIdx.x;
    const int lane = tid & 31;
    const int wave = tid >> 5;
    const int h    = blockIdx.y;
    const int b    = blockIdx.z;
    const int q0   = blockIdx.x * 64 + wave * 16;
    const int half = lane >> 4;
    const int lcol = lane & 15;
    const float scale = 0.08838834764831845f;   // 1/sqrt(128)

    __shared__ unsigned short Kl[2][64 * 136];   // [key][d], stride 136 (272B rows)
    __shared__ unsigned short Vl[2][128 * 72];   // [d][key], stride 72 (144B rows)
    __shared__ unsigned short Pl[4][16 * 72];    // per-wave 16x64 P tile
    unsigned short* pl = Pl[wave];

    const size_t khead = (size_t)b * S * D + (size_t)h * hd;      // + key*D + d
    const size_t vhead = (size_t)(b * H + h) * hd * S;            // + d*S + key

    // stage K tile (64x128: 1024 16B-chunks) + V tile (128x64: 1024 chunks);
    // 128 threads -> 8 + 8 async b128 per thread
    auto stage = [&](int buf, int key0) {
#pragma unroll
        for (int c = tid; c < 1024; c += 128) {
            int krow = c >> 4, kpart = c & 15;    // K: row=key(64), 16 chunks of 8 elems
            async_copy_b128(lds_off32(&Kl[buf][krow * 136 + kpart * 8]),
                            Kp + khead + (size_t)(key0 + krow) * D + kpart * 8);
            int vrow = c >> 3, vpart = c & 7;     // V: row=d(128), 8 chunks of 8 elems
            async_copy_b128(lds_off32(&Vl[buf][vrow * 72 + vpart * 8]),
                            Vt + vhead + (size_t)vrow * S + key0 + vpart * 8);
        }
    };

    // load Q fragments once (4 K-chunks of 32 over hd=128)
    const size_t qbase = ((size_t)b * S + q0) * D + (size_t)h * hd;
    v16bf qf[4];
#pragma unroll
    for (int kk = 0; kk < 4; ++kk)
        qf[kk] = load_frag_a(Qp + qbase + kk * 32, D, lane);

    v8f o[8];
#pragma unroll
    for (int nf = 0; nf < 8; ++nf) o[nf] = {};
    float m_i[8], l_i[8];
#pragma unroll
    for (int i = 0; i < 8; ++i) { m_i[i] = -1e30f; l_i[i] = 0.0f; }

    const int KT = S / 64;
    stage(0, 0);

    for (int kt = 0; kt < KT; ++kt) {
        const int cur  = kt & 1;
        if (kt + 1 < KT) {
            stage(cur ^ 1, (kt + 1) * 64);
            wait_async_le16();   // in-order completion -> current K/V tile landed
        } else {
            wait_async_le0();
        }
        __syncthreads();

        // S = Q K^T for a 16x64 tile; batch the 4 K-fragment loads per sub-tile
        v8f s[4];
#pragma unroll
        for (int j = 0; j < 4; ++j) {
            v16bf kf[4];
#pragma unroll
            for (int kk = 0; kk < 4; ++kk)
                kf[kk] = load_frag_b(&Kl[cur][(j * 16) * 136 + kk * 32], 136, lane);
            v8f acc = {};
#pragma unroll
            for (int kk = 0; kk < 4; ++kk)
                acc = __builtin_amdgcn_wmma_f32_16x16x32_bf16(
                    false, qf[kk], false, kf[kk], (short)0, acc, false, false);
            s[j] = acc;
        }

        // online softmax per row (rows = i + 8*half; cols across 16 lanes)
        float p[4][8];
#pragma unroll
        for (int i = 0; i < 8; ++i) {
            float v0 = s[0][i] * scale, v1 = s[1][i] * scale;
            float v2 = s[2][i] * scale, v3 = s[3][i] * scale;
            float rm = fmaxf(fmaxf(v0, v1), fmaxf(v2, v3));
#pragma unroll
            for (int off = 1; off < 16; off <<= 1)
                rm = fmaxf(rm, __shfl_xor(rm, off, 16));
            float mn    = fmaxf(m_i[i], rm);
            float alpha = __expf(m_i[i] - mn);
            float p0 = __expf(v0 - mn), p1 = __expf(v1 - mn);
            float p2 = __expf(v2 - mn), p3 = __expf(v3 - mn);
            float rs = p0 + p1 + p2 + p3;
#pragma unroll
            for (int off = 1; off < 16; off <<= 1)
                rs += __shfl_xor(rs, off, 16);
            l_i[i] = l_i[i] * alpha + rs;
            m_i[i] = mn;
#pragma unroll
            for (int nf = 0; nf < 8; ++nf) o[nf][i] *= alpha;
            p[0][i] = p0; p[1][i] = p1; p[2][i] = p2; p[3][i] = p3;
        }

        // transpose P through (wave-private) LDS into A-fragment layout.
        // LDS ops are in-order within a wave; fences stop compiler reordering.
        wave_fence();
#pragma unroll
        for (int j = 0; j < 4; ++j)
#pragma unroll
            for (int i = 0; i < 8; ++i)
                pl[(i + 8 * half) * 72 + j * 16 + lcol] = f32_to_bf16(p[j][i]);
        wave_fence();

        // O += P @ V (K=64 in two 32-chunks; V fragments from LDS, batched)
#pragma unroll
        for (int kk2 = 0; kk2 < 2; ++kk2) {
            v16bf pf = load_frag_a(pl + kk2 * 32, 72, lane);
#pragma unroll
            for (int ng = 0; ng < 2; ++ng) {
                v16bf vf[4];
#pragma unroll
                for (int u = 0; u < 4; ++u)
                    vf[u] = load_frag_b(&Vl[cur][((ng * 4 + u) * 16) * 72 + kk2 * 32], 72, lane);
#pragma unroll
                for (int u = 0; u < 4; ++u)
                    o[ng * 4 + u] = __builtin_amdgcn_wmma_f32_16x16x32_bf16(
                        false, pf, false, vf[u], (short)0, o[ng * 4 + u], false, false);
            }
        }
        __syncthreads();   // everyone done with Kl/Vl[cur] before overwrite next iter
    }

    // normalize and write fp32 output
#pragma unroll
    for (int nf = 0; nf < 8; ++nf)
#pragma unroll
        for (int i = 0; i < 8; ++i) {
            int row = q0 + i + 8 * half;
            mh[((size_t)b * S + row) * D + h * hd + nf * 16 + lcol] = o[nf][i] / l_i[i];
        }
}

// ---------------------------------------------------------------------------
// Fused residual + LayerNorm (torch semantics: std is ddof=1, no epsilon):
//   out = res + (x - mean) / std ; optionally also bf16 copy of out.
// ---------------------------------------------------------------------------
__global__ __launch_bounds__(256) void ln_residual_kernel(
    const float* __restrict__ x, const float* __restrict__ res,
    float* __restrict__ outf, unsigned short* __restrict__ outb, int D) {

    const int row  = blockIdx.x;
    const int tid  = threadIdx.x;
    const int lane = tid & 31;
    const int wave = tid >> 5;

    const float* xr = x + (size_t)row * D;
    float4 v = *(const float4*)(xr + tid * 4);
    float s  = v.x + v.y + v.z + v.w;
    float ss = v.x * v.x + v.y * v.y + v.z * v.z + v.w * v.w;
#pragma unroll
    for (int off = 1; off < 32; off <<= 1) {
        s  += __shfl_xor(s, off, 32);
        ss += __shfl_xor(ss, off, 32);
    }
    __shared__ float sbuf[8], ssbuf[8];
    if (lane == 0) { sbuf[wave] = s; ssbuf[wave] = ss; }
    __syncthreads();
    float tot = 0.0f, tot2 = 0.0f;
#pragma unroll
    for (int i = 0; i < 8; ++i) { tot += sbuf[i]; tot2 += ssbuf[i]; }

    float mean = tot / (float)D;
    float var  = (tot2 - tot * mean) / (float)(D - 1);
    float inv  = rsqrtf(var);

    const float* rr = res + (size_t)row * D;
    float4 r = *(const float4*)(rr + tid * 4);
    float4 o;
    o.x = r.x + (v.x - mean) * inv;
    o.y = r.y + (v.y - mean) * inv;
    o.z = r.z + (v.z - mean) * inv;
    o.w = r.w + (v.w - mean) * inv;
    *(float4*)(outf + (size_t)row * D + tid * 4) = o;
    if (outb) {
        uint2 ob;
        ob.x = (uint32_t)f32_to_bf16(o.x) | ((uint32_t)f32_to_bf16(o.y) << 16);
        ob.y = (uint32_t)f32_to_bf16(o.z) | ((uint32_t)f32_to_bf16(o.w) << 16);
        *(uint2*)(outb + (size_t)row * D + tid * 4) = ob;
    }
}

// ---------------------------------------------------------------------------
// Launch
// ---------------------------------------------------------------------------
extern "C" void kernel_launch(void* const* d_in, const int* in_sizes, int n_in,
                              void* d_out, int out_size, void* d_ws, size_t ws_size,
                              hipStream_t stream) {
    (void)in_sizes; (void)n_in; (void)out_size; (void)ws_size;

    const int B = 4, S = 2048, D = 1024, H = 8, hd = 128, Dff = 4096;
    const int M = B * S;

    const float* q  = (const float*)d_in[0];
    const float* k  = (const float*)d_in[1];
    const float* v  = (const float*)d_in[2];
    const float* Wq = (const float*)d_in[3];
    const float* bq = (const float*)d_in[4];
    const float* Wk = (const float*)d_in[5];
    const float* bk = (const float*)d_in[6];
    const float* Wv = (const float*)d_in[7];
    const float* bv = (const float*)d_in[8];
    const float* W1 = (const float*)d_in[9];
    const float* b1 = (const float*)d_in[10];
    const float* W2 = (const float*)d_in[11];
    const float* b2 = (const float*)d_in[12];
    float* out = (float*)d_out;

    // workspace carve-up
    char* wsp = (char*)d_ws;
    auto take = [&](size_t bytes) -> void* {
        void* p = (void*)wsp;
        wsp += (bytes + 255) & ~(size_t)255;
        return p;
    };
    unsigned short* qbf   = (unsigned short*)take((size_t)M * D * 2);
    unsigned short* kbf   = (unsigned short*)take((size_t)M * D * 2);
    unsigned short* vbf   = (unsigned short*)take((size_t)M * D * 2);
    unsigned short* WqT   = (unsigned short*)take((size_t)D * D * 2);
    unsigned short* WkT   = (unsigned short*)take((size_t)D * D * 2);
    unsigned short* WvT   = (unsigned short*)take((size_t)D * D * 2);
    unsigned short* W1T   = (unsigned short*)take((size_t)D * Dff * 2);
    unsigned short* W2T   = (unsigned short*)take((size_t)Dff * D * 2);
    unsigned short* Qp    = (unsigned short*)take((size_t)M * D * 2);
    unsigned short* Kp    = (unsigned short*)take((size_t)M * D * 2);
    unsigned short* Vp    = (unsigned short*)take((size_t)M * D * 2);
    unsigned short* Vt    = (unsigned short*)take((size_t)M * D * 2);
    float*          mh    = (float*)take((size_t)M * D * 4);
    float*          res1  = (float*)take((size_t)M * D * 4);
    unsigned short* res1b = (unsigned short*)take((size_t)M * D * 2);
    unsigned short* ff1   = (unsigned short*)take((size_t)M * Dff * 2);
    float*          ff2   = (float*)take((size_t)M * D * 4);

    const int nact = M * D;
    cvt_bf16_kernel<<<nact / 1024, 256, 0, stream>>>(q, qbf, nact);
    cvt_bf16_kernel<<<nact / 1024, 256, 0, stream>>>(k, kbf, nact);
    cvt_bf16_kernel<<<nact / 1024, 256, 0, stream>>>(v, vbf, nact);

    transpose_cvt_kernel<<<(D * D) / 256, 256, 0, stream>>>(Wq, WqT, D, D);
    transpose_cvt_kernel<<<(D * D) / 256, 256, 0, stream>>>(Wk, WkT, D, D);
    transpose_cvt_kernel<<<(D * D) / 256, 256, 0, stream>>>(Wv, WvT, D, D);
    transpose_cvt_kernel<<<(D * Dff) / 256, 256, 0, stream>>>(W1, W1T, D, Dff);
    transpose_cvt_kernel<<<(Dff * D) / 256, 256, 0, stream>>>(W2, W2T, Dff, D);

    // Projections (reference permutation: Q <- v, K <- q, V <- k)
    dim3 gproj(D / 128, M / 256);
    gemm_bf16_wmma_kernel<false, false, true><<<gproj, 256, 0, stream>>>(
        vbf, WqT, bq, nullptr, Qp, M, D, D);
    gemm_bf16_wmma_kernel<false, false, true><<<gproj, 256, 0, stream>>>(
        qbf, WkT, bk, nullptr, Kp, M, D, D);
    gemm_bf16_wmma_kernel<false, false, true><<<gproj, 256, 0, stream>>>(
        kbf, WvT, bv, nullptr, Vp, M, D, D);

    transpose_v_kernel<<<nact / 256, 256, 0, stream>>>(Vp, Vt, B, S, D, H, hd);

    attn_flash_kernel<<<dim3(S / 64, H, B), 128, 0, stream>>>(Qp, Kp, Vt, mh, S, D, H);

    ln_residual_kernel<<<M, 256, 0, stream>>>(mh, q, res1, res1b, D);

    gemm_bf16_wmma_kernel<true, false, true><<<dim3(Dff / 128, M / 256), 256, 0, stream>>>(
        res1b, W1T, b1, nullptr, ff1, M, Dff, D);
    gemm_bf16_wmma_kernel<false, true, false><<<dim3(D / 128, M / 256), 256, 0, stream>>>(
        ff1, W2T, b2, ff2, nullptr, M, D, Dff);

    ln_residual_kernel<<<M, 256, 0, stream>>>(ff2, res1, out, nullptr, D);
}
